// Mamba_71451075936340
// MI455X (gfx1250) — matrix operand order repaired and checked by reading, not compiled, not run on hardware
//
#include <hip/hip_runtime.h>
#include <math.h>

// ---------------- model constants ----------------
#define HIDN 1024
#define DIN  2048
#define NST  16
#define DTRK 64
#define XW   96        // DTR + 2*N
#define LSEQ 2048
#define NBAT 2

// ---------------- types ----------------
typedef __bf16        v16bf  __attribute__((ext_vector_type(16)));
typedef float         f32x8  __attribute__((ext_vector_type(8)));
typedef unsigned int  u32x4v __attribute__((ext_vector_type(4)));
typedef unsigned int  uint32x4 __attribute__((ext_vector_type(4)));
typedef int           int32x8  __attribute__((ext_vector_type(8)));
typedef int           int32x4  __attribute__((ext_vector_type(4)));

union FragU { v16bf bf; u32x4v q[2]; };

__device__ __forceinline__ unsigned short f2bf(float f) {
  unsigned int u = __float_as_uint(f);
  unsigned int r = 0x7FFFu + ((u >> 16) & 1u);   // round-to-nearest-even
  return (unsigned short)((u + r) >> 16);
}

// =====================================================================
// TDM: async 2D tile load (bf16) Global -> LDS with descriptor padding.
// tile = 32 elements (64B) x 128 rows; LDS rows padded 64B+16B = 80B,
// matching the 40-ushort-stride fragment layout.
// Issued once per wave that executes it (EXEC ignored); caller gates to
// wave 0 and synchronizes with s_wait_tensorcnt + workgroup barrier.
// =====================================================================
__device__ __forceinline__ void tdm_load_tile_bf16(
    unsigned lds_off, const unsigned short* gptr,
    unsigned k_rem, unsigned rows_rem, unsigned stride_elems)
{
  unsigned long long ga = (unsigned long long)(uintptr_t)gptr;

  uint32x4 g0;
  g0[0] = 1u;                                            // count=1 (valid user D#)
  g0[1] = lds_off;                                       // lds_addr (bytes)
  g0[2] = (unsigned)(ga & 0xFFFFFFFFull);                // global_addr[31:0]
  g0[3] = (unsigned)((ga >> 32) & 0x1FFFFFFull)          // global_addr[56:32]
        | (2u << 30);                                    // type = 2 ("image")

  int32x8 g1;
  // data_size=1 (2B) | pad_enable | pad_interval=3 (16 DW) | pad_amount=3 (4 DW)
  g1[0] = (1 << 16) | (1 << 20) | (3 << 22) | (3 << 25);
  unsigned td0 = k_rem;                                  // tensor_dim0 (elems)
  unsigned td1 = rows_rem;                               // tensor_dim1 (rows)
  g1[1] = (int)((td0 & 0xFFFFu) << 16);                  // [63:48] dim0 lo
  g1[2] = (int)(((td0 >> 16) & 0xFFFFu) | ((td1 & 0xFFFFu) << 16));
  g1[3] = (int)(((td1 >> 16) & 0xFFFFu) | (32u << 16));  // tile_dim0 = 32
  g1[4] = (int)128u;                                     // tile_dim1 = 128, tile_dim2 = 0
  unsigned long long s0 = stride_elems;                  // dim0 stride (48-bit)
  g1[5] = (int)(s0 & 0xFFFFFFFFull);
  g1[6] = (int)((s0 >> 32) & 0xFFFFull);                 // dim1_stride[15:0]=0
  g1[7] = 0;

  int32x4 z4; z4[0] = 0; z4[1] = 0; z4[2] = 0; z4[3] = 0;
  int32x8 z8; z8[0] = 0; z8[1] = 0; z8[2] = 0; z8[3] = 0;
  z8[4] = 0; z8[5] = 0; z8[6] = 0; z8[7] = 0;

  // 6-arg toolchain form: (g0, g1, g2, g3, g4, cpol)
  __builtin_amdgcn_tensor_load_to_lds(g0, g1, z4, z4, z8, 0);
}

// =====================================================================
// GEMM: C[M,N] = A[M,K] @ Bt[N,K]^T (+bias / softplus), A,Bt bf16,
// f32 WMMA accumulation; optional bf16 side output for GEMM chaining.
// Block = 256 threads = 8 wave32; tile 128x128; BK = 32.
// Double-buffered TDM staging: DMA for step k+1 is issued before
// computing step k; in-order TENSORcnt lets us wait <=2 (the next
// tile pair) so DMA latency hides behind the 8 WMMAs per step.
// =====================================================================
template<int MODE>   // 0: +bias, 1: softplus
__global__ __launch_bounds__(256)
void gemm_bf16_wmma(const unsigned short* __restrict__ A,
                    const unsigned short* __restrict__ Bt,
                    const float* __restrict__ bias,
                    float* __restrict__ C, unsigned short* __restrict__ Cbf,
                    int M, int N, int K, int lda, int ldbt, int ldc)
{
  __shared__ unsigned short a_sh [2][128][40];   // ping-pong, 80B rows (TDM padded)
  __shared__ unsigned short bt_sh[2][128][40];

  const int tid    = threadIdx.x;
  const int wave   = tid >> 5;
  const int lane   = tid & 31;
  const int half16 = lane >> 4;
  const int r      = lane & 15;

  const int m_blk = blockIdx.y * 128;
  const int n_blk = blockIdx.x * 128;

  const int wm = wave >> 1;        // 0..3 -> 32 rows each
  const int wn = wave & 1;         // 0..1 -> 64 cols each
  const int m_wave = wm * 32;
  const int n_wave = wn * 64;

  const unsigned lds_a [2] = { (unsigned)(uintptr_t)&a_sh [0][0][0],
                               (unsigned)(uintptr_t)&a_sh [1][0][0] };
  const unsigned lds_bt[2] = { (unsigned)(uintptr_t)&bt_sh[0][0][0],
                               (unsigned)(uintptr_t)&bt_sh[1][0][0] };
  const int n_rows_rem = N - n_blk;     // Bt rows available (OOB rows read 0)

  f32x8 acc[2][4];
  for (int i = 0; i < 2; ++i)
    for (int j = 0; j < 4; ++j)
      for (int e = 0; e < 8; ++e) acc[i][j][e] = 0.f;

  const int nk = K >> 5;     // all K are multiples of 32

  // ---- prologue: pre-issue tile pair 0 into buffer 0 ----
  if (wave == 0) {
    tdm_load_tile_bf16(lds_a[0],  &A [(size_t)m_blk * lda],
                       (unsigned)K, 128u, (unsigned)lda);
    tdm_load_tile_bf16(lds_bt[0], &Bt[(size_t)n_blk * ldbt],
                       (unsigned)K, (unsigned)n_rows_rem, (unsigned)ldbt);
  }

  for (int k = 0; k < nk; ++k) {
    const int cur = k & 1;
    if (wave == 0) {
      if (k + 1 < nk) {
        const int k0n = (k + 1) << 5;
        tdm_load_tile_bf16(lds_a[cur ^ 1],  &A [(size_t)m_blk * lda  + k0n],
                           (unsigned)(K - k0n), 128u, (unsigned)lda);
        tdm_load_tile_bf16(lds_bt[cur ^ 1], &Bt[(size_t)n_blk * ldbt + k0n],
                           (unsigned)(K - k0n), (unsigned)n_rows_rem,
                           (unsigned)ldbt);
        __builtin_amdgcn_s_wait_tensorcnt(2);   // pair k complete
      } else {
        __builtin_amdgcn_s_wait_tensorcnt(0);   // last pair complete
      }
    }
    __syncthreads();

    // ---- fragments per gfx1250 16-bit layouts ----
    FragU aF[2], bF[4];
#pragma unroll
    for (int i = 0; i < 2; ++i) {
      // A 16x32: lanes 0-15 K 0..7 & 16..23; lanes 16-31 K 8..15 & 24..31
      const unsigned short* ap = &a_sh[cur][m_wave + i * 16 + r][0];
      aF[i].q[0] = *reinterpret_cast<const u32x4v*>(ap + half16 * 8);
      aF[i].q[1] = *reinterpret_cast<const u32x4v*>(ap + 16 + half16 * 8);
    }
#pragma unroll
    for (int j = 0; j < 4; ++j) {
      // B 32x16: lanes 0-15 K 0..7 & 16..23; lanes 16-31 K 8..15 & 24..31, column = lane%16
      const unsigned short* bp = &bt_sh[cur][n_wave + j * 16 + r][half16 * 8];
      bF[j].q[0] = *reinterpret_cast<const u32x4v*>(bp);
      bF[j].q[1] = *reinterpret_cast<const u32x4v*>(bp + 16);
    }
#pragma unroll
    for (int i = 0; i < 2; ++i)
#pragma unroll
      for (int j = 0; j < 4; ++j)
        acc[i][j] = __builtin_amdgcn_wmma_f32_16x16x32_bf16(
            false, aF[i].bf, false, bF[j].bf, (short)0, acc[i][j],
            false, false);
    __syncthreads();   // readers done before buf `cur` is re-targeted
  }

  // ---- epilogue: D layout VGPR v -> M = v + 8*half, N = lane%16 ----
#pragma unroll
  for (int i = 0; i < 2; ++i) {
#pragma unroll
    for (int j = 0; j < 4; ++j) {
      int colg = n_blk + n_wave + j * 16 + r;
      if (colg >= N) continue;
#pragma unroll
      for (int v = 0; v < 8; ++v) {
        int rowg = m_blk + m_wave + i * 16 + v + 8 * half16;
        if (rowg >= M) continue;
        float val = acc[i][j][v];
        if (MODE == 0) {
          if (bias) val += bias[colg];
        } else {
          val = (val > 20.f) ? val : logf(1.f + __expf(val));  // softplus
        }
        C[(size_t)rowg * ldc + colg] = val;
        if (Cbf) Cbf[(size_t)rowg * ldc + colg] = f2bf(val);
      }
    }
  }
}

// =====================================================================
// Prep: f32 -> bf16 cast (x)
// =====================================================================
__global__ __launch_bounds__(256)
void cast_bf16_kernel(const float* __restrict__ in,
                      unsigned short* __restrict__ out, int n)
{
  int i = blockIdx.x * 256 + threadIdx.x;
  if (i < n) out[i] = f2bf(in[i]);
}

// =====================================================================
// Prep: w[K,N] f32 -> wT[N,K] bf16 (32x32 LDS tile transpose)
// =====================================================================
__global__ __launch_bounds__(256)
void transpose_cast_kernel(const float* __restrict__ in,
                           unsigned short* __restrict__ out, int K, int N)
{
  __shared__ float t[32][33];
  int k0 = blockIdx.y * 32, n0 = blockIdx.x * 32;
  int tx = threadIdx.x & 31, ty = threadIdx.x >> 5;   // ty: 0..7
#pragma unroll
  for (int i = 0; i < 32; i += 8) {
    int k = k0 + ty + i, n = n0 + tx;
    t[ty + i][tx] = (k < K && n < N) ? in[(size_t)k * N + n] : 0.f;
  }
  __syncthreads();
#pragma unroll
  for (int i = 0; i < 32; i += 8) {
    int n = n0 + ty + i, k = k0 + tx;
    if (n < N && k < K) out[(size_t)n * K + k] = f2bf(t[tx][ty + i]);
  }
}

// =====================================================================
// Depthwise causal conv (TW=4) + SiLU; writes f32 (for scan) + bf16
// (GEMM-A for x_dbl).  xc = xr[:, 0:DIN] (row stride 2*DIN).
// =====================================================================
__global__ __launch_bounds__(256)
void conv_silu_kernel(const float* __restrict__ xr,
                      const float* __restrict__ w_conv,
                      const float* __restrict__ b_conv,
                      float* __restrict__ u, unsigned short* __restrict__ u_bf)
{
  int idx = blockIdx.x * 256 + threadIdx.x;       // over NBAT*LSEQ*DIN
  int d   = idx & (DIN - 1);
  int row = idx >> 11;                            // DIN = 2^11
  int t   = row & (LSEQ - 1);
  float acc = b_conv[d];
#pragma unroll
  for (int k = 0; k < 4; ++k) {
    int tt = t - 3 + k;
    if (tt >= 0)
      acc = fmaf(w_conv[k * DIN + d],
                 xr[(size_t)(row - t + tt) * (2 * DIN) + d], acc);
  }
  float s = acc / (1.f + __expf(-acc));           // silu
  u[idx]    = s;
  u_bf[idx] = f2bf(s);
}

// =====================================================================
// Selective scan: one lane per (batch, d, n) chain -> 65536 lanes.
// h in a register; deltaA via v_exp per step; sum_n h*C via 4-step
// shfl_xor butterfly in each 16-lane group; silu(res) gating fused;
// writes bf16 directly (consumed as GEMM-A by the output projection).
// =====================================================================
__global__ __launch_bounds__(256)
void scan_kernel(const float* __restrict__ delta, const float* __restrict__ u,
                 const float* __restrict__ xdbl,  const float* __restrict__ xr,
                 const float* __restrict__ A_log, const float* __restrict__ Dp,
                 unsigned short* __restrict__ yz_bf)
{
  int tid = blockIdx.x * 256 + threadIdx.x;
  int n   = tid & (NST - 1);
  int dg  = tid >> 4;            // 0 .. NBAT*DIN-1
  int d   = dg & (DIN - 1);
  int bb  = dg >> 11;

  const float Aval = -__expf(A_log[d * NST + n]);
  const float Dv   = Dp[d];
  float h = 0.f;
  const size_t rbase = (size_t)bb * LSEQ;

  for (int t = 0; t < LSEQ; ++t) {
    const size_t row = rbase + t;
    if (n == 0 && (t & 15) == 0 && (t + 16) < LSEQ) {
      __builtin_prefetch(&delta[(row + 16) * DIN + d], 0, 1);
      __builtin_prefetch(&u[(row + 16) * DIN + d], 0, 1);
    }
    float dv = delta[row * DIN + d];
    float uv = u[row * DIN + d];
    float Bv = xdbl[row * XW + DTRK + n];
    float Cv = xdbl[row * XW + DTRK + NST + n];

    float a = __expf(dv * Aval);          // deltaA
    h = fmaf(a, h, dv * Bv * uv);         // h = a*h + delta*B*u

    float yv = h * Cv;                    // contract over n (16 lanes)
    yv += __shfl_xor(yv, 8, 32);
    yv += __shfl_xor(yv, 4, 32);
    yv += __shfl_xor(yv, 2, 32);
    yv += __shfl_xor(yv, 1, 32);

    if (n == 0) {
      float res = xr[row * (2 * DIN) + DIN + d];
      float g   = res / (1.f + __expf(-res));     // silu(res)
      yz_bf[row * DIN + d] = f2bf((yv + uv * Dv) * g);
    }
  }
}

// =====================================================================
// Host orchestration (graph-capture safe: only launches on `stream`).
// =====================================================================
extern "C" void kernel_launch(void* const* d_in, const int* in_sizes, int n_in,
                              void* d_out, int out_size, void* d_ws, size_t ws_size,
                              hipStream_t stream) {
  const float* x      = (const float*)d_in[0];
  const float* w_in   = (const float*)d_in[1];
  const float* b_in   = (const float*)d_in[2];
  const float* w_conv = (const float*)d_in[3];
  const float* b_conv = (const float*)d_in[4];
  const float* w_x    = (const float*)d_in[5];
  const float* w_dt   = (const float*)d_in[6];
  const float* A_log  = (const float*)d_in[7];
  const float* Dp     = (const float*)d_in[8];
  const float* w_out  = (const float*)d_in[9];
  float* out = (float*)d_out;

  const int BL = NBAT * LSEQ;   // 4096 rows

  // ---- workspace layout ----
  float* ws    = (float*)d_ws;
  float* xr    = ws;                              // [BL, 2*DIN] f32
  float* u     = xr    + (size_t)BL * 2 * DIN;    // [BL, DIN]   f32
  float* xdbl  = u     + (size_t)BL * DIN;        // [BL, 96]    f32
  float* delta = xdbl  + (size_t)BL * XW;         // [BL, DIN]   f32
  unsigned short* bfb    = (unsigned short*)(delta + (size_t)BL * DIN);
  unsigned short* x_bf   = bfb;                                   // [BL,1024]
  unsigned short* w_inT  = x_bf   + (size_t)BL * HIDN;            // [4096,1024]
  unsigned short* w_xT   = w_inT  + (size_t)(2 * DIN) * HIDN;     // [96,2048]
  unsigned short* w_dtT  = w_xT   + (size_t)XW * DIN;             // [2048,64]
  unsigned short* w_outT = w_dtT  + (size_t)DIN * DTRK;           // [1024,2048]
  unsigned short* u_bf   = w_outT + (size_t)HIDN * DIN;           // [BL,2048]
  unsigned short* xdblbf = u_bf   + (size_t)BL * DIN;             // [BL,96]
  unsigned short* yz_bf  = xdblbf + (size_t)BL * XW;              // [BL,2048]

  dim3 blk(256);

  // ---- prep: casts / weight transposes (bf16, [N,K] row-major) ----
  cast_bf16_kernel<<<dim3((BL * HIDN + 255) / 256), blk, 0, stream>>>(
      x, x_bf, BL * HIDN);
  transpose_cast_kernel<<<dim3((2 * DIN) / 32, HIDN / 32), blk, 0, stream>>>(
      w_in, w_inT, HIDN, 2 * DIN);
  transpose_cast_kernel<<<dim3(XW / 32, DIN / 32), blk, 0, stream>>>(
      w_x, w_xT, DIN, XW);
  transpose_cast_kernel<<<dim3(DIN / 32, DTRK / 32), blk, 0, stream>>>(
      w_dt, w_dtT, DTRK, DIN);
  transpose_cast_kernel<<<dim3(HIDN / 32, DIN / 32), blk, 0, stream>>>(
      w_out, w_outT, DIN, HIDN);

  // 1) xr = x @ w_in + b_in        [4096,1024]x[1024,4096]
  gemm_bf16_wmma<0><<<dim3((2 * DIN) / 128, BL / 128), blk, 0, stream>>>(
      x_bf, w_inT, b_in, xr, nullptr, BL, 2 * DIN, HIDN, HIDN, HIDN, 2 * DIN);

  // 2) u = silu(causal_conv(xc))   (f32 + bf16)
  conv_silu_kernel<<<dim3((BL * DIN) / 256), blk, 0, stream>>>(
      xr, w_conv, b_conv, u, u_bf);

  // 3) x_dbl = u @ w_x             [4096,2048]x[2048,96]  (f32 + bf16)
  gemm_bf16_wmma<0><<<dim3(1, BL / 128), blk, 0, stream>>>(
      u_bf, w_xT, nullptr, xdbl, xdblbf, BL, XW, DIN, DIN, DIN, XW);

  // 4) delta = softplus(dt @ w_dt) [4096,64]x[64,2048]  (dt = xdbl[:, :64] bf16)
  gemm_bf16_wmma<1><<<dim3(DIN / 128, BL / 128), blk, 0, stream>>>(
      xdblbf, w_dtT, nullptr, delta, nullptr, BL, DIN, DTRK, XW, DTRK, DIN);

  // 5) selective scan + D skip + silu(res) gating -> yz (bf16)
  scan_kernel<<<dim3((NBAT * DIN * NST) / 256), blk, 0, stream>>>(
      delta, u, xdbl, xr, A_log, Dp, yz_bf);

  // 6) out = yz @ w_out            [4096,2048]x[2048,1024]
  gemm_bf16_wmma<0><<<dim3(HIDN / 128, BL / 128), blk, 0, stream>>>(
      yz_bf, w_outT, nullptr, out, nullptr, BL, HIDN, DIN, DIN, DIN, HIDN);
}
